// EncodeProcessDecode_8727373545623
// MI455X (gfx1250) — matrix operand order repaired and checked by reading, not compiled, not run on hardware
//
#include <hip/hip_runtime.h>
#include <cstdint>

typedef __attribute__((ext_vector_type(16))) _Float16 v16h;
typedef __attribute__((ext_vector_type(8)))  _Float16 v8h;
typedef __attribute__((ext_vector_type(8)))  float    v8f;

#define LDH 136   // 128 + 8 halves padding -> 272B row stride in LDS, conflict-free

struct GemmArgs {
  const _Float16* src0; const int* idx0;
  const _Float16* src1; const int* idx1;
  const _Float16* src2; const int* idx2;
  int nsrc; int Ktot; int M;
  const _Float16* wt;    // [128][Ktot] f16, N-major rows (transposed weights)
  const float* bias;     // [128]
};

struct Epi {
  const float* lng; const float* lnb;  // layernorm gamma/beta (MODE 1)
  float* res; int add_res;             // f32 master: if add_res, v += res; then res = v
  _Float16* outh;                      // f16 output [M][128]
  float* scat; const int* scat_idx;    // fused segment-sum: atomicAdd(scat[idx[row]][col], v)
};

// ---- CDNA5 async global->LDS DMA (ASYNCcnt-tracked), per-lane 16B ----------
__device__ __forceinline__ void async_g2l_b128(uint32_t lds_byte_off, const void* gaddr) {
  asm volatile("global_load_async_to_lds_b128 %0, %1, off"
               :: "v"(lds_byte_off), "v"(gaddr) : "memory");
}
__device__ __forceinline__ void wait_async_le8() {
  asm volatile("s_wait_asynccnt 0x8" ::: "memory");
}
__device__ __forceinline__ void wait_async_0() {
  asm volatile("s_wait_asynccnt 0x0" ::: "memory");
}

// ---------------------------------------------------------------------------
// Generic fused MLP layer: out = epilog( concat(src0..srcK) @ Wt + bias )
// 256 threads = 8 waves. Block tile: 128 rows x 128 cols.
// Weights (shared by all waves) are async-DMA'd to LDS, double-buffered per
// K-chunk; A fragments (per-wave private, optionally index-gathered) stream
// straight from global (L2-resident features).
// ---------------------------------------------------------------------------
template<int MODE>   // 0 = ReLU -> f16 ; 1 = LayerNorm (+residual, +scatter)
__global__ __launch_bounds__(256) void k_mlp(GemmArgs ga, Epi ep) {
  __shared__ _Float16 sW[2][128 * LDH];   // double-buffered weight chunk (2 x 34KB)

  const int tid  = threadIdx.x;
  const int wave = tid >> 5;
  const int lane = tid & 31;
  const int ln   = lane & 15;
  const bool up  = lane >= 16;
  const int rbase = blockIdx.x * 128 + wave * 16;

  v8f acc[8] = {};

  // A-fragment (16x32 f16): lanes 0-15 row=ln K{0..7,16..23}; 16-31 row=ln K{8..15,24..31}
  const int koffA = up ? 8 : 0;
  // B-fragment (32x16 f16): lanes 0-15 col=ln K{0..7,16..23}; 16-31 col=ln K{8..15,24..31}
  const int koffB = up ? 8 : 0;

  // per-lane gathered A row (clamped for partial last tile)
  int growA = rbase + ln;
  if (growA >= ga.M) growA = ga.M - 1;

  // weight staging: thread t DMAs 128B of row (t>>1) at half-offset (t&1)*64
  const int wrow = tid >> 1;
  const int sho  = (tid & 1) * 64;

  auto stage = [&](int buf, int s) {
    const _Float16* wsrc = ga.wt + (size_t)wrow * ga.Ktot + s * 128 + sho;
    const uint32_t lds0 = (uint32_t)(uintptr_t)(&sW[buf][wrow * LDH + sho]);
#pragma unroll
    for (int i = 0; i < 8; ++i)
      async_g2l_b128(lds0 + i * 16, wsrc + i * 8);
  };

  stage(0, 0);

  for (int s = 0; s < ga.nsrc; ++s) {
    const bool more = (s + 1) < ga.nsrc;
    if (more) stage((s + 1) & 1, s + 1);          // prefetch next chunk's weights
    if (more) wait_async_le8(); else wait_async_0();
    __syncthreads();                               // chunk-s weights visible to all waves

    const _Float16* sp = (s == 0) ? ga.src0 : ((s == 1) ? ga.src1 : ga.src2);
    const int* gi      = (s == 0) ? ga.idx0 : ((s == 1) ? ga.idx1 : ga.idx2);
    const int srow = gi ? gi[growA] : growA;
    const _Float16* ap = sp + (size_t)srow * 128;
    const _Float16* wb = &sW[s & 1][0];

#pragma unroll
    for (int kb = 0; kb < 128; kb += 32) {
      const v8h alo = *(const v8h*)(ap + kb + koffA);
      const v8h ahi = *(const v8h*)(ap + kb + 16 + koffA);
      v16h a;
#pragma unroll
      for (int i = 0; i < 8; ++i) { a[i] = alo[i]; a[8 + i] = ahi[i]; }
#pragma unroll
      for (int nt = 0; nt < 8; ++nt) {
        const _Float16* wp = wb + (nt * 16 + ln) * LDH + kb + koffB;
        const v8h blo = *(const v8h*)(wp);
        const v8h bhi = *(const v8h*)(wp + 16);
        v16h b;
#pragma unroll
        for (int i = 0; i < 8; ++i) { b[i] = blo[i]; b[8 + i] = bhi[i]; }
        acc[nt] = __builtin_amdgcn_wmma_f32_16x16x32_f16(
            false, a, false, b, (short)0, acc[nt], false, false);
      }
    }
    if (more) __syncthreads();   // all waves done reading buf[s&1] before it is re-staged
  }

  float bcol[8];
#pragma unroll
  for (int nt = 0; nt < 8; ++nt) bcol[nt] = ga.bias[nt * 16 + ln];

  if constexpr (MODE == 0) {
    // bias + ReLU -> f16
#pragma unroll
    for (int nt = 0; nt < 8; ++nt) {
      const int col = nt * 16 + ln;
#pragma unroll
      for (int r = 0; r < 8; ++r) {
        const int row = rbase + r + (up ? 8 : 0);
        if (row < ga.M) {
          float v = acc[nt][r] + bcol[nt];
          v = fmaxf(v, 0.f);
          ep.outh[(size_t)row * 128 + col] = (_Float16)v;
        }
      }
    }
  } else {
    // bias + LayerNorm (+residual f32 master, f16 mirror, optional scatter-add)
    float gcol[8], zcol[8];
#pragma unroll
    for (int nt = 0; nt < 8; ++nt) {
      const int col = nt * 16 + ln;
      gcol[nt] = ep.lng[col];
      zcol[nt] = ep.lnb[col];
    }
#pragma unroll
    for (int r = 0; r < 8; ++r) {
      float s1 = 0.f, s2 = 0.f;
#pragma unroll
      for (int nt = 0; nt < 8; ++nt) {
        const float v = acc[nt][r] + bcol[nt];
        s1 += v; s2 += v * v;
      }
      // one row lives in one 16-lane half-wave; reduce within it
      s1 += __shfl_xor(s1, 1, 32);  s2 += __shfl_xor(s2, 1, 32);
      s1 += __shfl_xor(s1, 2, 32);  s2 += __shfl_xor(s2, 2, 32);
      s1 += __shfl_xor(s1, 4, 32);  s2 += __shfl_xor(s2, 4, 32);
      s1 += __shfl_xor(s1, 8, 32);  s2 += __shfl_xor(s2, 8, 32);
      const float mean = s1 * (1.f / 128.f);
      const float var  = s2 * (1.f / 128.f) - mean * mean;
      const float rs   = rsqrtf(var + 1e-5f);

      const int row = rbase + r + (up ? 8 : 0);
      const bool ok = row < ga.M;
      int di = 0;
      if (ep.scat_idx && ok) di = ep.scat_idx[row];
#pragma unroll
      for (int nt = 0; nt < 8; ++nt) {
        const int col = nt * 16 + ln;
        float v = acc[nt][r] + bcol[nt];
        v = (v - mean) * rs * gcol[nt] + zcol[nt];
        if (ep.res && ok) {
          if (ep.add_res) v += ep.res[(size_t)row * 128 + col];
          ep.res[(size_t)row * 128 + col] = v;
        }
        if (ok) ep.outh[(size_t)row * 128 + col] = (_Float16)v;
        if (ep.scat && ok) atomicAdd(ep.scat + (size_t)di * 128 + col, v);
      }
    }
  }
}

// --------------------------- small helper kernels ---------------------------

// Transpose f32 W[K][128] -> f16 Wt[128][K] (N-major, K contiguous)
__global__ void k_transpose(const float* __restrict__ w, int K, _Float16* __restrict__ dst) {
  const int t = blockIdx.x * blockDim.x + threadIdx.x;
  if (t >= 128 * K) return;
  const int n = t / K, k = t - n * K;
  dst[t] = (_Float16)w[(size_t)k * 128 + n];
}

// First layer with odd din (30 or 7): scalar FMA, negligible FLOPs
__global__ void k_in_layer(const float* __restrict__ x, int M, int din,
                           const float* __restrict__ w, const float* __restrict__ b,
                           _Float16* __restrict__ out) {
  const int t = blockIdx.x * blockDim.x + threadIdx.x;
  if (t >= M * 128) return;
  const int row = t >> 7, col = t & 127;
  float acc = b[col];
  for (int k = 0; k < din; ++k)
    acc = fmaf(x[(size_t)row * din + k], w[(size_t)k * 128 + col], acc);
  out[t] = (_Float16)fmaxf(acc, 0.f);
}

// Decoder head: [M,128] f16 @ [128,3] f32 + b -> f32 out
__global__ void k_dec_final(const _Float16* __restrict__ h, int M,
                            const float* __restrict__ w, const float* __restrict__ b,
                            float* __restrict__ out) {
  const int t = blockIdx.x * blockDim.x + threadIdx.x;
  if (t >= M * 3) return;
  const int row = t / 3, j = t - row * 3;
  float acc = b[j];
  for (int k = 0; k < 128; ++k)
    acc = fmaf((float)h[(size_t)row * 128 + k], w[k * 3 + j], acc);
  out[t] = acc;
}

__global__ void k_zero(float* p, int n) {
  const int t = blockIdx.x * blockDim.x + threadIdx.x;
  if (t < n) p[t] = 0.f;
}

__global__ void k_f32_to_f16(const float* __restrict__ a, _Float16* __restrict__ o, int n) {
  const int t = blockIdx.x * blockDim.x + threadIdx.x;
  if (t < n) o[t] = (_Float16)a[t];
}

// ---------------------------------------------------------------------------

extern "C" void kernel_launch(void* const* d_in, const int* in_sizes, int n_in,
                              void* d_out, int out_size, void* d_ws, size_t ws_size,
                              hipStream_t stream) {
  (void)n_in; (void)out_size; (void)ws_size;
  const float* node_x    = (const float*)d_in[0];
  const float* edge_attr = (const float*)d_in[1];
  const int*   eidx      = (const int*)d_in[2];
  const int N = in_sizes[0] / 30;
  const int E = in_sizes[1] / 7;
  const int* esrc = eidx;       // edge_index[0] = x_j source
  const int* edst = eidx + E;   // edge_index[1] = x_i / aggregation target

  auto F = [&](int i) { return (const float*)d_in[i]; };

  // ---- workspace carve ----
  uintptr_t base = (uintptr_t)d_ws;
  auto carve = [&](size_t bytes) -> void* {
    base = (base + 255) & ~(uintptr_t)255;
    void* p = (void*)base; base += bytes; return p;
  };

  // Param leaf order (jax pytree, dict keys sorted):
  //  3..8   dec  (W0,b0,W1,b1,W2,b2)
  //  9..16  enc_edge (W0,b0,W1,b1,W2,b2,g,b)
  // 17..24  enc_node (W0,b0,W1,b1,W2,b2,g,b)
  // 25+16s  step s: msg W0,b0,W1,b1,W2,b2,g,b ; upd W0,b0,W1,b1,W2,b2,g,b
  struct TW { int idx; int K; };
  TW tws[24]; int ntw = 0;
  tws[ntw++] = {3, 128};   // dec W0
  tws[ntw++] = {5, 128};   // dec W1
  tws[ntw++] = {11, 128};  // enc_edge W1
  tws[ntw++] = {13, 128};  // enc_edge W2
  tws[ntw++] = {19, 128};  // enc_node W1
  tws[ntw++] = {21, 128};  // enc_node W2
  for (int s = 0; s < 3; ++s) {
    const int b = 25 + s * 16;
    tws[ntw++] = {b + 0, 384};   // msg W0
    tws[ntw++] = {b + 2, 128};   // msg W1
    tws[ntw++] = {b + 4, 128};   // msg W2
    tws[ntw++] = {b + 8, 256};   // upd W0
    tws[ntw++] = {b + 10, 128};  // upd W1
    tws[ntw++] = {b + 12, 128};  // upd W2
  }
  _Float16* wtp[24];
  for (int i = 0; i < 24; ++i) wtp[i] = (_Float16*)carve((size_t)128 * tws[i].K * 2);

  _Float16* xh   = (_Float16*)carve((size_t)N * 128 * 2);
  float*    xf   = (float*)   carve((size_t)N * 128 * 4);
  _Float16* eah  = (_Float16*)carve((size_t)E * 128 * 2);
  float*    eaf  = (float*)   carve((size_t)E * 128 * 4);
  _Float16* t0   = (_Float16*)carve((size_t)E * 128 * 2);
  _Float16* t1   = (_Float16*)carve((size_t)E * 128 * 2);
  float*    aggf = (float*)   carve((size_t)N * 128 * 4);
  _Float16* aggh = (_Float16*)carve((size_t)N * 128 * 2);

  // ---- weight prep (runs every call; tiny) ----
  for (int i = 0; i < 24; ++i) {
    const int total = 128 * tws[i].K;
    k_transpose<<<(total + 255) / 256, 256, 0, stream>>>(F(tws[i].idx), tws[i].K, wtp[i]);
  }

  auto gemm_relu = [&](const _Float16* s0, const int* i0,
                       const _Float16* s1, const int* i1,
                       const _Float16* s2, const int* i2,
                       int nsrc, int M, const _Float16* wt, const float* bias,
                       _Float16* out) {
    GemmArgs ga{s0, i0, s1, i1, s2, i2, nsrc, nsrc * 128, M, wt, bias};
    Epi ep{nullptr, nullptr, nullptr, 0, out, nullptr, nullptr};
    k_mlp<0><<<(M + 127) / 128, 256, 0, stream>>>(ga, ep);
  };
  auto gemm_ln = [&](const _Float16* s0, const int* i0,
                     const _Float16* s1, const int* i1,
                     const _Float16* s2, const int* i2,
                     int nsrc, int M, const _Float16* wt, const float* bias,
                     const float* g, const float* bb, float* res, int add_res,
                     _Float16* outh, float* scat, const int* scat_idx) {
    GemmArgs ga{s0, i0, s1, i1, s2, i2, nsrc, nsrc * 128, M, wt, bias};
    Epi ep{g, bb, res, add_res, outh, scat, scat_idx};
    k_mlp<1><<<(M + 127) / 128, 256, 0, stream>>>(ga, ep);
  };

  const int Nt = N * 128, Et = E * 128;

  // ---- encoders ----
  k_in_layer<<<(Nt + 255) / 256, 256, 0, stream>>>(node_x, N, 30, F(17), F(18), t0);
  gemm_relu(t0, nullptr, nullptr, nullptr, nullptr, nullptr, 1, N, wtp[4], F(20), t1);
  gemm_ln  (t1, nullptr, nullptr, nullptr, nullptr, nullptr, 1, N, wtp[5], F(22),
            F(23), F(24), xf, 0, xh, nullptr, nullptr);

  k_in_layer<<<(Et + 255) / 256, 256, 0, stream>>>(edge_attr, E, 7, F(9), F(10), t0);
  gemm_relu(t0, nullptr, nullptr, nullptr, nullptr, nullptr, 1, E, wtp[2], F(12), t1);
  gemm_ln  (t1, nullptr, nullptr, nullptr, nullptr, nullptr, 1, E, wtp[3], F(14),
            F(15), F(16), eaf, 0, eah, nullptr, nullptr);

  // ---- processor steps ----
  for (int s = 0; s < 3; ++s) {
    const int b = 25 + s * 16;
    const int wb = 6 + s * 6;
    k_zero<<<(Nt + 255) / 256, 256, 0, stream>>>(aggf, Nt);

    // msg: concat[x_i, x_j, ea] -> fused as 3 gathered K-chunks
    gemm_relu(xh, edst, xh, esrc, eah, nullptr, 3, E, wtp[wb + 0], F(b + 1), t0);
    gemm_relu(t0, nullptr, nullptr, nullptr, nullptr, nullptr, 1, E, wtp[wb + 1], F(b + 3), t1);
    // LN + ea residual + fused scatter-add of new ea into aggf[dst]
    gemm_ln  (t1, nullptr, nullptr, nullptr, nullptr, nullptr, 1, E, wtp[wb + 2], F(b + 5),
              F(b + 6), F(b + 7), eaf, 1, eah, aggf, edst);

    k_f32_to_f16<<<(Nt + 255) / 256, 256, 0, stream>>>(aggf, aggh, Nt);

    // upd: concat[x, agg]
    gemm_relu(xh, nullptr, aggh, nullptr, nullptr, nullptr, 2, N, wtp[wb + 3], F(b + 9), t0);
    gemm_relu(t0, nullptr, nullptr, nullptr, nullptr, nullptr, 1, N, wtp[wb + 4], F(b + 11), t1);
    gemm_ln  (t1, nullptr, nullptr, nullptr, nullptr, nullptr, 1, N, wtp[wb + 5], F(b + 13),
              F(b + 14), F(b + 15), xf, 1, xh, nullptr, nullptr);
  }

  // ---- decoder ----
  gemm_relu(xh, nullptr, nullptr, nullptr, nullptr, nullptr, 1, N, wtp[0], F(4), t0);
  gemm_relu(t0, nullptr, nullptr, nullptr, nullptr, nullptr, 1, N, wtp[1], F(6), t1);
  k_dec_final<<<(N * 3 + 255) / 256, 256, 0, stream>>>(t1, N, F(7), F(8), (float*)d_out);
}
